// PhaseAttention_85521388798334
// MI455X (gfx1250) — matrix operand order repaired and checked by reading, not compiled, not run on hardware
//
#include <hip/hip_runtime.h>
#include <hip/hip_bf16.h>
#include <math.h>

typedef __attribute__((ext_vector_type(16))) __bf16 v16bf;
typedef __attribute__((ext_vector_type(8)))  __bf16 v8bf;
typedef __attribute__((ext_vector_type(4)))  __bf16 v4bf;
typedef __attribute__((ext_vector_type(8)))  float  v8f;

#define SEQ_L   96
#define DMODEL  256
#define NHEADS  8
#define DHEAD   32
#define WIN     16

// A-operand (16x32 bf16, MxK): lane 0-15 -> row=lane, K = kbase+0..15
//                              lane 16-31 -> row=lane-16, K = kbase+16..31
static __device__ __forceinline__ v16bf load_a_frag(const __bf16* base, int ld, int row,
                                                    int kbase, int lane) {
  int koff = (lane & 16) ? 16 : 0;
  const __bf16* p = base + row * ld + kbase + koff;
  v8bf lo = *(const v8bf*)(p);
  v8bf hi = *(const v8bf*)(p + 8);
  v16bf f;
#pragma unroll
  for (int i = 0; i < 8; ++i) { f[i] = lo[i]; f[i + 8] = hi[i]; }
  return f;
}

// B-operand (32x16 bf16, KxN) from N-major storage (row = N column, contiguous K):
// lanes 0-15: K = kbase+0..15 ; lanes 16-31: K = kbase+16..31
static __device__ __forceinline__ v16bf load_b_frag(const __bf16* base, int ld, int row,
                                                    int kbase, int lane) {
  int koff = (lane & 16) ? 16 : 0;
  return *(const v16bf*)(base + row * ld + kbase + koff);
}

static __device__ __forceinline__ v8f wmma_bf16(v16bf a, v16bf b, v8f c) {
  return __builtin_amdgcn_wmma_f32_16x16x32_bf16(false, a, false, b, (short)0, c, false, false);
}

// C/D layout: VGPR r, lanes 0-15 -> (row=r, col=lane); lanes 16-31 -> (row=r+8, col=lane-16)
static __device__ __forceinline__ void store_tile(__bf16* base, int ld, int rowbase,
                                                  int colbase, v8f d, int lane) {
  int col = colbase + (lane & 15);
  int r0 = (lane & 16) ? 8 : 0;
#pragma unroll
  for (int r = 0; r < 8; ++r) base[(rowbase + r0 + r) * ld + col] = (__bf16)d[r];
}

__global__ void convert_weights_kernel(const float* __restrict__ Wq, const float* __restrict__ Wk,
                                       const float* __restrict__ Wv, const float* __restrict__ Wo,
                                       __bf16* __restrict__ WqT, __bf16* __restrict__ WkT,
                                       __bf16* __restrict__ WvT, __bf16* __restrict__ WoT) {
  int idx = blockIdx.x * blockDim.x + threadIdx.x;
  if (idx >= DMODEL * DMODEL) return;
  int r = idx >> 8;          // output column of W  -> row of W^T
  int c = idx & 255;         // contraction index
  int src = c * DMODEL + r;
  WqT[idx] = (__bf16)Wq[src];
  WkT[idx] = (__bf16)Wk[src];
  WvT[idx] = (__bf16)Wv[src];
  WoT[idx] = (__bf16)Wo[src];
}

// LDS layout (bytes):
//   xb  [96][256] bf16 : 0      .. 49152   (reused as attention-output "ob")
//   qb  [96][256] bf16 : 49152  .. 98304
//   kb  [96][256] bf16 : 98304  .. 147456
//   vtb [256][96] bf16 : 147456 .. 196608  (V transposed: [dh_col][key])
//   sst 8x[16][96] bf16: 196608 .. 221184  (per-wave softmax staging)
//   biasb [8][96] f32  : 221184 .. 224256
#define SMEM_BYTES 224256

__global__ __launch_bounds__(256)
void phase_attn_kernel(const float* __restrict__ Z, const float* __restrict__ rel_bias,
                       const __bf16* __restrict__ WqT, const __bf16* __restrict__ WkT,
                       const __bf16* __restrict__ WvT, const __bf16* __restrict__ WoT,
                       float* __restrict__ Out) {
  extern __shared__ char smem[];
  __bf16* xb  = (__bf16*)(smem);
  __bf16* qb  = (__bf16*)(smem + 49152);
  __bf16* kb  = (__bf16*)(smem + 98304);
  __bf16* vtb = (__bf16*)(smem + 147456);
  __bf16* sst_all = (__bf16*)(smem + 196608);
  float*  biasb   = (float*)(smem + 221184);

  const int tid  = threadIdx.x;
  const int lane = tid & 31;
  const int w    = tid >> 5;          // wave id == head id
  const int n    = blockIdx.x;        // sequence id (B*C = 1024)

  // ---- Stage 0: load Z sequence -> bf16 in LDS; cache rel_bias ----
  const float4* zsrc = (const float4*)(Z + (size_t)n * (SEQ_L * DMODEL));
#pragma unroll
  for (int t = 0; t < 24; ++t) {
    int f = tid + t * 256;            // 6144 float4 total
    float4 val = zsrc[f];
    v4bf tmp;
    tmp[0] = (__bf16)val.x; tmp[1] = (__bf16)val.y;
    tmp[2] = (__bf16)val.z; tmp[3] = (__bf16)val.w;
    *(v4bf*)(xb + f * 4) = tmp;
  }
  for (int t = tid; t < NHEADS * SEQ_L; t += 256) biasb[t] = rel_bias[t];
  __syncthreads();

  const int cb0  = w * 32;
  const int cb1  = w * 32 + 16;
  const int rsel = lane & 15;
  const int r0   = (lane & 16) ? 8 : 0;

  // ---- Stage 1: fused Q/K/V projections. Wave w -> cols [32w,32w+32) == head w ----
  for (int m = 0; m < 6; ++m) {
    v8f q0 = {}, q1 = {}, k0 = {}, k1 = {}, v0 = {}, v1 = {};
    for (int ks = 0; ks < 8; ++ks) {
      v16bf a = load_a_frag(xb, DMODEL, m * 16 + rsel, ks * 32, lane);
      q0 = wmma_bf16(a, load_b_frag(WqT, DMODEL, cb0 + rsel, ks * 32, lane), q0);
      q1 = wmma_bf16(a, load_b_frag(WqT, DMODEL, cb1 + rsel, ks * 32, lane), q1);
      k0 = wmma_bf16(a, load_b_frag(WkT, DMODEL, cb0 + rsel, ks * 32, lane), k0);
      k1 = wmma_bf16(a, load_b_frag(WkT, DMODEL, cb1 + rsel, ks * 32, lane), k1);
      v0 = wmma_bf16(a, load_b_frag(WvT, DMODEL, cb0 + rsel, ks * 32, lane), v0);
      v1 = wmma_bf16(a, load_b_frag(WvT, DMODEL, cb1 + rsel, ks * 32, lane), v1);
    }
    store_tile(qb, DMODEL, m * 16, cb0, q0, lane);
    store_tile(qb, DMODEL, m * 16, cb1, q1, lane);
    store_tile(kb, DMODEL, m * 16, cb0, k0, lane);
    store_tile(kb, DMODEL, m * 16, cb1, k1, lane);
    {  // V stored transposed: per-lane 8 contiguous bf16 -> one 16B LDS store
      v8bf tv;
#pragma unroll
      for (int r = 0; r < 8; ++r) tv[r] = (__bf16)v0[r];
      *(v8bf*)(vtb + (cb0 + rsel) * SEQ_L + m * 16 + r0) = tv;
#pragma unroll
      for (int r = 0; r < 8; ++r) tv[r] = (__bf16)v1[r];
      *(v8bf*)(vtb + (cb1 + rsel) * SEQ_L + m * 16 + r0) = tv;
    }
  }
  __syncthreads();

  // ---- Stage 2: windowed attention; wave w processes head w entirely ----
  const int h = w;
  __bf16* sst = sst_all + w * (16 * SEQ_L);
  const float scl = 0.17677669529663687f;  // 1/sqrt(32)
  for (int m = 0; m < 6; ++m) {
    v8f acc[6];
    v16bf aq = load_a_frag(qb, DMODEL, m * 16 + rsel, h * DHEAD, lane);  // K = Dh = 32 exact
#pragma unroll
    for (int kt = 0; kt < 6; ++kt) {
      v8f z = {};
      v16bf bk = load_b_frag(kb, DMODEL, kt * 16 + rsel, h * DHEAD, lane);  // lane = key col
      acc[kt] = wmma_bf16(aq, bk, z);
    }
    // scale + circular relative bias + +/-16 circular window mask
#pragma unroll
    for (int kt = 0; kt < 6; ++kt) {
#pragma unroll
      for (int r = 0; r < 8; ++r) {
        int qrow = m * 16 + r0 + r;
        int key  = kt * 16 + rsel;
        int dd   = qrow - key;
        int ad   = dd < 0 ? -dd : dd;
        int dc   = ad < (SEQ_L - ad) ? ad : (SEQ_L - ad);
        int rel  = dd < 0 ? dd + SEQ_L : dd;
        float b  = (dc <= WIN) ? biasb[h * SEQ_L + rel] : -3.0e38f;
        acc[kt][r] = acc[kt][r] * scl + b;
      }
    }
    // row softmax: a row = 6 regs/lane across 16 lanes of one half-wave
#pragma unroll
    for (int r = 0; r < 8; ++r) {
      float mx = -3.0e38f;
#pragma unroll
      for (int kt = 0; kt < 6; ++kt) mx = fmaxf(mx, acc[kt][r]);
#pragma unroll
      for (int s = 1; s < 16; s <<= 1) mx = fmaxf(mx, __shfl_xor(mx, s, 32));
      float sum = 0.f;
#pragma unroll
      for (int kt = 0; kt < 6; ++kt) { float p = __expf(acc[kt][r] - mx); acc[kt][r] = p; sum += p; }
#pragma unroll
      for (int s = 1; s < 16; s <<= 1) sum += __shfl_xor(sum, s, 32);
      float inv = 1.0f / sum;
#pragma unroll
      for (int kt = 0; kt < 6; ++kt)
        sst[(r0 + r) * SEQ_L + kt * 16 + rsel] = (__bf16)(acc[kt][r] * inv);
    }
    // attn @ V : A = probs [16 x 96] from staging, B = vT (contiguous over keys)
    v16bf ap0 = load_a_frag(sst, SEQ_L, rsel, 0,  lane);
    v16bf ap1 = load_a_frag(sst, SEQ_L, rsel, 32, lane);
    v16bf ap2 = load_a_frag(sst, SEQ_L, rsel, 64, lane);
#pragma unroll
    for (int nt = 0; nt < 2; ++nt) {
      v8f o = {};
      o = wmma_bf16(ap0, load_b_frag(vtb, SEQ_L, h * DHEAD + nt * 16 + rsel, 0,  lane), o);
      o = wmma_bf16(ap1, load_b_frag(vtb, SEQ_L, h * DHEAD + nt * 16 + rsel, 32, lane), o);
      o = wmma_bf16(ap2, load_b_frag(vtb, SEQ_L, h * DHEAD + nt * 16 + rsel, 64, lane), o);
      store_tile(xb /* reuse as ob */, DMODEL, m * 16, h * DHEAD + nt * 16, o, lane);
    }
  }
  __syncthreads();

  // ---- Stage 3: output projection ob @ Wo -> fp32 global ----
  float* dst = Out + (size_t)n * (SEQ_L * DMODEL);
  for (int m = 0; m < 6; ++m) {
    v8f o0 = {}, o1 = {};
    for (int ks = 0; ks < 8; ++ks) {
      v16bf a = load_a_frag(xb, DMODEL, m * 16 + rsel, ks * 32, lane);
      o0 = wmma_bf16(a, load_b_frag(WoT, DMODEL, cb0 + rsel, ks * 32, lane), o0);
      o1 = wmma_bf16(a, load_b_frag(WoT, DMODEL, cb1 + rsel, ks * 32, lane), o1);
    }
#pragma unroll
    for (int r = 0; r < 8; ++r) {
      dst[(m * 16 + r0 + r) * DMODEL + cb0 + rsel] = o0[r];
      dst[(m * 16 + r0 + r) * DMODEL + cb1 + rsel] = o1[r];
    }
  }
}

extern "C" void kernel_launch(void* const* d_in, const int* in_sizes, int n_in,
                              void* d_out, int out_size, void* d_ws, size_t ws_size,
                              hipStream_t stream) {
  (void)in_sizes; (void)n_in; (void)out_size; (void)ws_size;
  const float* Z  = (const float*)d_in[0];
  const float* Wq = (const float*)d_in[1];
  const float* Wk = (const float*)d_in[2];
  const float* Wv = (const float*)d_in[3];
  const float* Wo = (const float*)d_in[4];
  const float* rb = (const float*)d_in[5];

  __bf16* WqT = (__bf16*)d_ws;                 // 4 x 64K bf16 = 512 KB in d_ws
  __bf16* WkT = WqT + DMODEL * DMODEL;
  __bf16* WvT = WkT + DMODEL * DMODEL;
  __bf16* WoT = WvT + DMODEL * DMODEL;

  hipFuncSetAttribute((const void*)phase_attn_kernel,
                      hipFuncAttributeMaxDynamicSharedMemorySize, SMEM_BYTES);

  convert_weights_kernel<<<256, 256, 0, stream>>>(Wq, Wk, Wv, Wo, WqT, WkT, WvT, WoT);
  phase_attn_kernel<<<1024, 256, SMEM_BYTES, stream>>>(Z, rb, WqT, WkT, WvT, WoT,
                                                       (float*)d_out);
}
